// ECForGraphTCN_84774064488721
// MI455X (gfx1250) — matrix operand order repaired and checked by reading, not compiled, not run on hardware
//
#include <hip/hip_runtime.h>
#include <hip/hip_bf16.h>
#include <math.h>

typedef __attribute__((ext_vector_type(16))) _Float16 v16h;
typedef __attribute__((ext_vector_type(8)))  _Float16 v8h;
typedef __attribute__((ext_vector_type(8)))  float    v8f;

#define NN 50000
#define EE 800000
#define NTILES_N (NN/16)   // 3125
#define NTILES_E (EE/16)   // 50000
#define ALPHA_C 0.5f
#define EPS_C   1e-3f

// ---- swizzled f16 weight buffer offsets (in halfs) ----
enum {
  OFF_ENW1 = 0,        // 32x64 (K padded 14->32)
  OFF_ENW2 = 2048,     // 64x32
  OFF_EEW1 = 4096,     // 32x64 (K padded 4->32)
  OFF_EEW2 = 6144,     // 64x32
  OFF_RW1  = 8192,     // 4 x 96x64  (6144 each)
  OFF_RW2  = 32768,    // 4 x 64x64  (4096 each)
  OFF_RW3  = 49152,    // 4 x 64x32  (2048 each)
  OFF_OW1  = 57344,    // 4 x 64x64
  OFF_OW2  = 73728,    // 4 x 64x64
  OFF_OW3  = 90112,    // 4 x 64x32
  OFF_WW1  = 98304,    // 192x64
  OFF_WW2  = 110592,   // 64x64
  WB_HALFS = 114688
};

// ---------------------------------------------------------------------------
// WMMA tile helpers (16-row tile per wave, f16 in / f32 accumulate)
// A fragment (16x32 f16): lanes 0-15 -> M=lane, K = k0+0..7 & k0+16..23
//                         lanes16-31 -> M=lane-16, K = k0+8..15 & k0+24..31
// B fragment pre-swizzled so each lane loads 16 contiguous halfs.
// C layout: vgpr r -> M = r + 8*(lane>=16), N = lane&15.
// ---------------------------------------------------------------------------
__device__ __forceinline__ v16h make_v16(v8h lo, v8h hi) {
  v16h a;
#pragma unroll
  for (int i = 0; i < 8; ++i) { a[i] = lo[i]; a[i + 8] = hi[i]; }
  return a;
}

__device__ __forceinline__ v8f gemm_acc(const _Float16* A, int K,
                                        const _Float16* Wsw, int ntiles, int nt, int lane) {
  const int grp = lane >> 4;
  const int m   = lane & 15;
  v8f acc = {};
  for (int kt = 0; kt < (K >> 5); ++kt) {
    const _Float16* ap = A + m * K + kt * 32 + grp * 8;
    v16h a = make_v16(*(const v8h*)ap, *(const v8h*)(ap + 16));
    const v8h* bp = (const v8h*)(Wsw + ((kt * ntiles + nt) * 32 + lane) * 16);
    v16h b = make_v16(bp[0], bp[1]);
    acc = __builtin_amdgcn_wmma_f32_16x16x32_f16(false, a, false, b, (short)0, acc,
                                                 false, false);
  }
  return acc;
}

__device__ __forceinline__ void gemm_relu_to_lds(const _Float16* A, int K,
                                                 const _Float16* Wsw, const float* bias,
                                                 _Float16* Out, int Nout, int lane) {
  const int grp = lane >> 4;
  const int n   = lane & 15;
  const int ntiles = Nout >> 4;
  for (int nt = 0; nt < ntiles; ++nt) {
    v8f acc = gemm_acc(A, K, Wsw, ntiles, nt, lane);
    float bv = bias ? bias[nt * 16 + n] : 0.f;
#pragma unroll
    for (int r = 0; r < 8; ++r) {
      float v = acc[r] + bv;
      v = v > 0.f ? v : 0.f;
      Out[(r + grp * 8) * Nout + nt * 16 + n] = (_Float16)v;
    }
  }
}

// ---------------------------------------------------------------------------
// Weight conversion: f32 row-major [K x Nout] -> f16 fragment-swizzled, K padded
// ---------------------------------------------------------------------------
__global__ void __launch_bounds__(256)
swz_weights(const float* __restrict__ src, _Float16* __restrict__ dst,
            int Ksrc, int Kpad, int Nout, int nmats) {
  int per = Kpad * Nout;
  int total = per * nmats;
  for (int id = blockIdx.x * blockDim.x + threadIdx.x; id < total;
       id += gridDim.x * blockDim.x) {
    int mat = id / per;
    int r = id - mat * per;
    int i = r & 15;
    int lane = (r >> 4) & 31;
    int tile = r >> 9;
    int ntiles = Nout >> 4;
    int kt = tile / ntiles;
    int nt = tile - kt * ntiles;
    int grp = lane >> 4;
    int n = nt * 16 + (lane & 15);
    int k = kt * 32 + grp * 8 + (i & 7) + ((i >> 3) << 4);
    float v = (k < Ksrc) ? src[((size_t)mat * Ksrc + k) * Nout + n] : 0.f;
    dst[id] = (_Float16)v;
  }
}

__global__ void __launch_bounds__(256)
fill_zero(float* __restrict__ p, int n) {
  for (int i = blockIdx.x * blockDim.x + threadIdx.x; i < n; i += gridDim.x * blockDim.x)
    p[i] = 0.f;
}

__global__ void __launch_bounds__(256)
copy_f32(const float* __restrict__ s, float* __restrict__ d, int n) {
  for (int i = blockIdx.x * blockDim.x + threadIdx.x; i < n; i += gridDim.x * blockDim.x)
    d[i] = s[i];
}

// ---------------------------------------------------------------------------
// Node encoder: h = relu(relu(x @ W1) @ W2), x [N,14] (K padded to 32)
// ---------------------------------------------------------------------------
__global__ void __launch_bounds__(128)
encode_nodes(const float* __restrict__ x, const _Float16* __restrict__ wb,
             float* __restrict__ h) {
  __shared__ __align__(16) _Float16 sh[4][16 * 32 + 16 * 64];
  const int lane = threadIdx.x & 31, wid = threadIdx.x >> 5;
  const int tile = blockIdx.x * 4 + wid;
  if (tile >= NTILES_N) return;
  _Float16* A0 = sh[wid];
  _Float16* A1 = A0 + 16 * 32;
  for (int idx = lane; idx < 16 * 32; idx += 32) {
    int r = idx >> 5, c = idx & 31;
    A0[idx] = (_Float16)((c < 14) ? x[((size_t)tile * 16 + r) * 14 + c] : 0.f);
  }
  __builtin_amdgcn_wave_barrier();
  gemm_relu_to_lds(A0, 32, wb + OFF_ENW1, nullptr, A1, 64, lane);
  __builtin_amdgcn_wave_barrier();
  const int grp = lane >> 4, n = lane & 15;
  for (int nt = 0; nt < 2; ++nt) {
    v8f acc = gemm_acc(A1, 64, wb + OFF_ENW2, 2, nt, lane);
#pragma unroll
    for (int r = 0; r < 8; ++r) {
      float v = acc[r]; v = v > 0.f ? v : 0.f;
      h[((size_t)tile * 16 + r + grp * 8) * 32 + nt * 16 + n] = v;
    }
  }
}

// ---------------------------------------------------------------------------
// Edge encoder: e = relu(relu(ea @ W1) @ W2), ea [E,4] (K padded to 32)
// ---------------------------------------------------------------------------
__global__ void __launch_bounds__(128)
encode_edges(const float* __restrict__ ea, const _Float16* __restrict__ wb,
             float* __restrict__ e) {
  __shared__ __align__(16) _Float16 sh[4][16 * 32 + 16 * 64];
  const int lane = threadIdx.x & 31, wid = threadIdx.x >> 5;
  const int tile = blockIdx.x * 4 + wid;   // exact: 50000 tiles / 4 waves
  _Float16* A0 = sh[wid];
  _Float16* A1 = A0 + 16 * 32;
  for (int idx = lane; idx < 16 * 32; idx += 32) {
    int r = idx >> 5, c = idx & 31;
    A0[idx] = (_Float16)((c < 4) ? ea[((size_t)tile * 16 + r) * 4 + c] : 0.f);
  }
  __builtin_amdgcn_wave_barrier();
  gemm_relu_to_lds(A0, 32, wb + OFF_EEW1, nullptr, A1, 64, lane);
  __builtin_amdgcn_wave_barrier();
  const int grp = lane >> 4, n = lane & 15;
  for (int nt = 0; nt < 2; ++nt) {
    v8f acc = gemm_acc(A1, 64, wb + OFF_EEW2, 2, nt, lane);
#pragma unroll
    for (int r = 0; r < 8; ++r) {
      float v = acc[r]; v = v > 0.f ? v : 0.f;
      e[((size_t)tile * 16 + r + grp * 8) * 32 + nt * 16 + n] = v;
    }
  }
}

// ---------------------------------------------------------------------------
// Per-layer edge message: rel-MLP([h[dst],h[src],e]) -> e_new;
// atomic segment-sum into agg[dst]; e = 0.5e + 0.5e_new; stash f16 embed.
// ---------------------------------------------------------------------------
__global__ void __launch_bounds__(128)
edge_msg(const float* __restrict__ h, float* __restrict__ e,
         const int* __restrict__ eidx, const _Float16* __restrict__ wb,
         const float* __restrict__ rb1, const float* __restrict__ rb2,
         const float* __restrict__ rb3, float* __restrict__ agg,
         _Float16* __restrict__ emb, int l) {
  __shared__ __align__(16) _Float16 sh[4][16 * 96 + 16 * 64];
  __shared__ int sidx[4][32];
  const int lane = threadIdx.x & 31, wid = threadIdx.x >> 5;
  const int tile = blockIdx.x * 4 + wid;
  const int eb = tile * 16;
  _Float16* A0 = sh[wid];
  _Float16* A1 = A0 + 16 * 96;
  // lanes 0-15: src ids; lanes 16-31: dst ids
  sidx[wid][lane] = (lane < 16) ? eidx[eb + lane] : eidx[EE + eb + (lane - 16)];
  __builtin_amdgcn_wave_barrier();
  for (int idx = lane; idx < 16 * 96; idx += 32) {
    int r = idx / 96, c = idx - r * 96;
    float v;
    if (c < 32)       v = h[(size_t)sidx[wid][16 + r] * 32 + c];        // h[dst]
    else if (c < 64)  v = h[(size_t)sidx[wid][r] * 32 + (c - 32)];      // h[src]
    else              v = e[((size_t)eb + r) * 32 + (c - 64)];          // e
    A0[idx] = (_Float16)v;
  }
  __builtin_amdgcn_wave_barrier();
  gemm_relu_to_lds(A0, 96, wb + OFF_RW1 + l * 6144, rb1 + l * 64, A1, 64, lane);
  __builtin_amdgcn_wave_barrier();
  gemm_relu_to_lds(A1, 64, wb + OFF_RW2 + l * 4096, rb2 + l * 64, A0, 64, lane);
  __builtin_amdgcn_wave_barrier();
  const int grp = lane >> 4, n = lane & 15;
  for (int nt = 0; nt < 2; ++nt) {
    v8f acc = gemm_acc(A0, 64, wb + OFF_RW3 + l * 2048, 2, nt, lane);
    float bv = rb3[l * 32 + nt * 16 + n];
#pragma unroll
    for (int r = 0; r < 8; ++r) {
      int M = r + grp * 8, col = nt * 16 + n;
      size_t eg = (size_t)eb + M;
      float enew = acc[r] + bv;               // linear last layer
      float er = ALPHA_C * e[eg * 32 + col] + (1.f - ALPHA_C) * enew;
      e[eg * 32 + col] = er;
      emb[((size_t)l * EE + eg) * 32 + col] = (_Float16)er;
      atomicAdd(&agg[(size_t)sidx[wid][16 + M] * 32 + col], enew);
    }
  }
}

// ---------------------------------------------------------------------------
// Per-layer node update: obj-MLP([h, agg]) -> h_new; h = 0.5h + 0.5h_new
// ---------------------------------------------------------------------------
__global__ void __launch_bounds__(128)
node_upd(float* __restrict__ h, const float* __restrict__ agg,
         const _Float16* __restrict__ wb, const float* __restrict__ ob1,
         const float* __restrict__ ob2, const float* __restrict__ ob3, int l) {
  __shared__ __align__(16) _Float16 sh[4][16 * 64 + 16 * 64];
  const int lane = threadIdx.x & 31, wid = threadIdx.x >> 5;
  const int tile = blockIdx.x * 4 + wid;
  if (tile >= NTILES_N) return;
  const int nb = tile * 16;
  _Float16* A0 = sh[wid];
  _Float16* A1 = A0 + 16 * 64;
  for (int idx = lane; idx < 16 * 64; idx += 32) {
    int r = idx >> 6, c = idx & 63;
    float v = (c < 32) ? h[((size_t)nb + r) * 32 + c]
                       : agg[((size_t)nb + r) * 32 + (c - 32)];
    A0[idx] = (_Float16)v;
  }
  __builtin_amdgcn_wave_barrier();
  gemm_relu_to_lds(A0, 64, wb + OFF_OW1 + l * 4096, ob1 + l * 64, A1, 64, lane);
  __builtin_amdgcn_wave_barrier();
  gemm_relu_to_lds(A1, 64, wb + OFF_OW2 + l * 4096, ob2 + l * 64, A0, 64, lane);
  __builtin_amdgcn_wave_barrier();
  const int grp = lane >> 4, n = lane & 15;
  for (int nt = 0; nt < 2; ++nt) {
    v8f acc = gemm_acc(A0, 64, wb + OFF_OW3 + l * 2048, 2, nt, lane);
    float bv = ob3[l * 32 + nt * 16 + n];
#pragma unroll
    for (int r = 0; r < 8; ++r) {
      size_t ng = (size_t)nb + r + grp * 8;
      float hnew = acc[r] + bv;
      h[ng * 32 + nt * 16 + n] =
          ALPHA_C * h[ng * 32 + nt * 16 + n] + (1.f - ALPHA_C) * hnew;
    }
  }
}

// ---------------------------------------------------------------------------
// Final edge scorer: MLP([h[src],h[dst],emb0..3]) -> sigmoid weight
// ---------------------------------------------------------------------------
__global__ void __launch_bounds__(128)
edge_score(const float* __restrict__ h, const _Float16* __restrict__ emb,
           const int* __restrict__ eidx, const _Float16* __restrict__ wb,
           const float* __restrict__ b1, const float* __restrict__ b2,
           const float* __restrict__ W3, const float* __restrict__ b3,
           float* __restrict__ wout) {
  __shared__ __align__(16) _Float16 sh[4][16 * 192 + 16 * 64 + 16 * 64];
  __shared__ int sidx[4][32];
  const int lane = threadIdx.x & 31, wid = threadIdx.x >> 5;
  const int tile = blockIdx.x * 4 + wid;
  const int eb = tile * 16;
  _Float16* A0 = sh[wid];
  _Float16* A1 = A0 + 16 * 192;
  _Float16* A2 = A1 + 16 * 64;
  sidx[wid][lane] = (lane < 16) ? eidx[eb + lane] : eidx[EE + eb + (lane - 16)];
  __builtin_amdgcn_wave_barrier();
  for (int idx = lane; idx < 16 * 192; idx += 32) {
    int r = idx / 192, c = idx - r * 192;
    _Float16 v;
    if (c < 32)       v = (_Float16)h[(size_t)sidx[wid][r] * 32 + c];          // h[src]
    else if (c < 64)  v = (_Float16)h[(size_t)sidx[wid][16 + r] * 32 + (c - 32)]; // h[dst]
    else {
      int li = (c - 64) >> 5, cc = (c - 64) & 31;
      v = emb[((size_t)li * EE + eb + r) * 32 + cc];
    }
    A0[idx] = v;
  }
  __builtin_amdgcn_wave_barrier();
  gemm_relu_to_lds(A0, 192, wb + OFF_WW1, b1, A1, 64, lane);
  __builtin_amdgcn_wave_barrier();
  gemm_relu_to_lds(A1, 64, wb + OFF_WW2, b2, A2, 64, lane);
  __builtin_amdgcn_wave_barrier();
  if (lane < 16) {                       // Nout == 1: 16-lane dot product
    float s = b3[0];
#pragma unroll 16
    for (int k = 0; k < 64; ++k) s += (float)A2[lane * 64 + k] * W3[k];
    float sig = 1.f / (1.f + expf(-s));
    wout[eb + lane] = EPS_C + (1.f - 2.f * EPS_C) * sig;
  }
}

// ---------------------------------------------------------------------------
extern "C" void kernel_launch(void* const* d_in, const int* in_sizes, int n_in,
                              void* d_out, int out_size, void* d_ws, size_t ws_size,
                              hipStream_t stream) {
  const float* x     = (const float*)d_in[0];
  const int*   eidx  = (const int*)d_in[1];
  const float* eattr = (const float*)d_in[2];
  const float* rb1 = (const float*)d_in[8];
  const float* rb2 = (const float*)d_in[10];
  const float* rb3 = (const float*)d_in[12];
  const float* ob1 = (const float*)d_in[14];
  const float* ob2 = (const float*)d_in[16];
  const float* ob3 = (const float*)d_in[18];
  const float* wb1 = (const float*)d_in[20];
  const float* wb2 = (const float*)d_in[22];
  const float* wW3 = (const float*)d_in[23];
  const float* wb3 = (const float*)d_in[24];

  _Float16* wb  = (_Float16*)d_ws;                      // 229 KB swizzled f16 weights
  float*    h   = (float*)((char*)d_ws + 262144);       // [N,32]
  float*    agg = h + (size_t)NN * 32;                  // [N,32]
  float*    e   = agg + (size_t)NN * 32;                // [E,32]
  _Float16* emb = (_Float16*)(e + (size_t)EE * 32);     // [4,E,32] f16

  auto swz = [&](const void* src, int off, int Ksrc, int Kpad, int Nout, int nm) {
    int total = Kpad * Nout * nm;
    swz_weights<<<dim3((total + 255) / 256), dim3(256), 0, stream>>>(
        (const float*)src, wb + off, Ksrc, Kpad, Nout, nm);
  };
  swz(d_in[3],  OFF_ENW1, 14, 32, 64, 1);
  swz(d_in[4],  OFF_ENW2, 64, 64, 32, 1);
  swz(d_in[5],  OFF_EEW1,  4, 32, 64, 1);
  swz(d_in[6],  OFF_EEW2, 64, 64, 32, 1);
  swz(d_in[7],  OFF_RW1,  96, 96, 64, 4);
  swz(d_in[9],  OFF_RW2,  64, 64, 64, 4);
  swz(d_in[11], OFF_RW3,  64, 64, 32, 4);
  swz(d_in[13], OFF_OW1,  64, 64, 64, 4);
  swz(d_in[15], OFF_OW2,  64, 64, 64, 4);
  swz(d_in[17], OFF_OW3,  64, 64, 32, 4);
  swz(d_in[19], OFF_WW1, 192, 192, 64, 1);
  swz(d_in[21], OFF_WW2,  64, 64, 64, 1);

  encode_nodes<<<dim3((NTILES_N + 3) / 4), dim3(128), 0, stream>>>(x, wb, h);
  encode_edges<<<dim3(NTILES_E / 4), dim3(128), 0, stream>>>(eattr, wb, e);

  for (int l = 0; l < 4; ++l) {
    fill_zero<<<dim3(1024), dim3(256), 0, stream>>>(agg, NN * 32);
    edge_msg<<<dim3(NTILES_E / 4), dim3(128), 0, stream>>>(h, e, eidx, wb, rb1, rb2,
                                                           rb3, agg, emb, l);
    node_upd<<<dim3((NTILES_N + 3) / 4), dim3(128), 0, stream>>>(h, agg, wb, ob1, ob2,
                                                                 ob3, l);
  }

  float* out_w = (float*)d_out;               // [E]
  float* out_h = out_w + EE;                  // [N,32]
  float* out_e = out_h + (size_t)NN * 32;     // [E,32]
  edge_score<<<dim3(NTILES_E / 4), dim3(128), 0, stream>>>(h, emb, eidx, wb, wb1, wb2,
                                                           wW3, wb3, out_w);
  copy_f32<<<dim3(1024), dim3(256), 0, stream>>>(h, out_h, NN * 32);
  copy_f32<<<dim3(4096), dim3(256), 0, stream>>>(e, out_e, EE * 32);
}
